// AllInOneLayer_70806830842551
// MI455X (gfx1250) — matrix operand order repaired and checked by reading, not compiled, not run on hardware
//
#include <hip/hip_runtime.h>
#include <math.h>

// ---------------------------------------------------------------------------
// AllInOneLayer for MI455X (gfx1250, wave32).
// All dense layers run on V_WMMA_F32_16X16X32_F16 (f32 accumulate).
// Weights are pre-packed once per launch into the wave32 WMMA B-fragment
// layout; activations are staged row-major in LDS (f16) and loaded into the
// A-fragment layout in contiguous K order (k = 16*hi + j), matching the packed B.
// Hot-path SiLU/sigmoid/softmax-normalization use v_rcp_f32 (1-ulp) instead
// of IEEE division: the GEMMs are already f16, and the exact-division
// expansion (~10 VALU ops/element) dominated the epilogue otherwise.
// ---------------------------------------------------------------------------

#define NN 20000   // nodes
#define NE 320000  // edges
#define NB 800     // graphs
#define DS 128
#define DV 15

typedef __attribute__((ext_vector_type(16))) _Float16 v16h;
typedef __attribute__((ext_vector_type(8)))  float    v8f;

__device__ __forceinline__ float fastrcp(float x){ return __builtin_amdgcn_rcpf(x); }
__device__ __forceinline__ float siluf(float x){ return x * fastrcp(1.f + __expf(-x)); }
__device__ __forceinline__ float sigmf(float x){ return fastrcp(1.f + __expf(-x)); }

__device__ __forceinline__ void atomicMaxF(float* a, float v){
  if (v >= 0.f) atomicMax((int*)a, __float_as_int(v));
  else          atomicMin((unsigned int*)a, (unsigned int)__float_as_int(v));
}

// One wave computes a 16-row GEMM tile: Out = act(A[16xK] @ W[KxN] + bias).
// A: LDS f16, row stride AS halfs (AS % 8 == 0), K = 32*KT, rows zero padded.
// Wpk: packed f16 B-fragments: tile (kt,nt) at ((kt*NTtot+nt)*32+lane)*16,
//      element j of lane L = W[kt*32 + (L>>4)*16 + j][nt*16 + (L&15)].
// Writes f16 (OutH, stride OS) and/or f32 (OutF, stride OFS).
__device__ __forceinline__ void wmma_gemm16(
    const _Float16* A, int AS, int KT,
    const _Float16* Wpk, int NTtot, int nt0, int ntCnt,
    const float* bias, bool act_silu,
    _Float16* OutH, int OS, float* OutF, int OFS)
{
  const int lane = threadIdx.x & 31;
  const int n  = lane & 15;
  const int hi = lane >> 4;
  const _Float16* arow = A + (lane & 15) * AS + hi * 16;
  for (int nt = 0; nt < ntCnt; ++nt) {
    v8f c = {0.f,0.f,0.f,0.f,0.f,0.f,0.f,0.f};
    for (int kt = 0; kt < KT; ++kt) {
      v16h a, b;
      const _Float16* ap = arow + kt * 32;
      // A fragment (contiguous K order, matching B): halves 0..15 = K[hi*16 .. hi*16+15]
      *((float4*)&a)       = *((const float4*)(ap));
      *(((float4*)&a) + 1) = *((const float4*)(ap + 8));
      b = *((const v16h*)(Wpk + ((size_t)(kt * NTtot + nt0 + nt) * 32 + lane) * 16));
      c = __builtin_amdgcn_wmma_f32_16x16x32_f16(false, a, false, b, (short)0, c, false, false);
    }
#pragma unroll
    for (int g = 0; g < 8; ++g) {
      int m   = hi * 8 + g;            // C layout: lanes16-31 hold M=8+g
      int col = (nt0 + nt) * 16 + n;
      float v = c[g] + bias[col];
      if (act_silu) v = siluf(v);
      int lc = nt * 16 + n;
      if (OutH) OutH[m * OS + lc] = (_Float16)v;
      if (OutF) OutF[m * OFS + lc] = v;
    }
  }
}

// ------------------------- prep kernels ------------------------------------
__global__ void k_fill(float* p, float v, int n){
  int i = blockIdx.x * blockDim.x + threadIdx.x;
  if (i < n) p[i] = v;
}

__global__ void k_padbias(const float* s, float* d, int n, int npad){
  int i = blockIdx.x * blockDim.x + threadIdx.x;
  if (i < npad) d[i] = (i < n) ? s[i] : 0.f;
}

// Pack row-major W[K][N] (leading dim ldw) into WMMA B fragments.
__global__ void k_pack(const float* W, _Float16* dst, int K, int N, int ldw,
                       int KT, int NTtot, int nt0, int NTcnt)
{
  int idx = blockIdx.x * blockDim.x + threadIdx.x;
  int total = KT * NTcnt * 512;
  if (idx >= total) return;
  int j    = idx & 15;
  int lane = (idx >> 4) & 31;
  int t    = idx >> 9;
  int ntl  = t % NTcnt;
  int kt   = t / NTcnt;
  int nt   = nt0 + ntl;
  int hi = lane >> 4, n = lane & 15;
  int k  = kt * 32 + hi * 16 + j;  // B fragment K striping (32x16 f16 tile)
  int wc = ntl * 16 + n;           // column relative to this weight
  float v = 0.f;
  if (k < K && wc < N) v = W[(size_t)k * ldw + wc];
  dst[((size_t)(kt * NTtot + nt) * 32 + lane) * 16 + j] = (_Float16)v;
}

// ----------------- pass 0: per-node MLP U = mlp2([s,v], node) --------------
__global__ void k_node_u(const float* s, const float* v,
                         const _Float16* pkW1, const float* b1,
                         const _Float16* pkW2, const float* b2, float* U)
{
  __shared__ _Float16 sA[2][16*160];
  __shared__ _Float16 sB[2][16*256];
  __shared__ float    sF[2][16*128];
  int wave = threadIdx.x >> 5, lane = threadIdx.x & 31;
  int n0 = (blockIdx.x * 2 + wave) * 16;
  _Float16* A = sA[wave]; _Float16* B = sB[wave]; float* F = sF[wave];
  for (int idx = lane; idx < 16*160; idx += 32) {
    int m = idx / 160, c = idx % 160;
    float val = 0.f;
    if      (c < 128) val = s[(size_t)(n0+m)*128 + c];
    else if (c < 143) val = v[(size_t)(n0+m)*15 + (c-128)];
    A[m*160 + c] = (_Float16)val;
  }
  __syncthreads();
  wmma_gemm16(A, 160, 5, pkW1, 16, 0, 16, b1, true,  B, 256, nullptr, 0);
  __syncthreads();
  wmma_gemm16(B, 256, 8, pkW2,  8, 0,  8, b2, false, nullptr, 0, F, 128);
  __syncthreads();
  for (int idx = lane; idx < 16*128; idx += 32) {
    int m = idx >> 7, c = idx & 127;
    U[(size_t)(n0+m)*128 + c] = F[m*128 + c];
  }
}

// ----------------- pass 1: attention logits + segment max ------------------
__global__ void k_attn(const float* s, const float* v,
                       const float* s_edge, const float* v_edge, const int* ei,
                       const _Float16* pkW1, const float* b1pad,
                       const float* W2, const float* b2,
                       float* RAW, float* M)
{
  __shared__ _Float16 sA[2][16*448];
  __shared__ _Float16 sH[2][16*144];
  __shared__ float    sPart[2][2][16];
  __shared__ int      sRow[2][16], sCol[2][16];
  int wave = threadIdx.x >> 5, lane = threadIdx.x & 31;
  int e0 = (blockIdx.x * 2 + wave) * 16;
  if (lane < 16) { sRow[wave][lane] = ei[e0+lane]; sCol[wave][lane] = ei[NE + e0 + lane]; }
  __syncthreads();
  _Float16* A = sA[wave];
  for (int idx = lane; idx < 16*448; idx += 32) {
    int m = idx / 448, c = idx % 448;
    int r = sRow[wave][m], cl = sCol[wave][m];
    float val = 0.f;
    if      (c < 128) val = s[(size_t)r*128 + c];
    else if (c < 256) val = s[(size_t)cl*128 + (c-128)];
    else if (c < 384) val = s_edge[(size_t)(e0+m)*128 + (c-256)];
    else if (c < 399) val = v[(size_t)r*15 + (c-384)];
    else if (c < 414) val = v[(size_t)cl*15 + (c-399)];
    else if (c < 429) val = v_edge[(size_t)(e0+m)*15 + (c-414)];
    A[m*448 + c] = (_Float16)val;
  }
  __syncthreads();
  for (int h = 0; h < 4; ++h) {
    wmma_gemm16(A, 448, 14, pkW1, 36, 9*h, 9, b1pad, true, sH[wave], 144, nullptr, 0);
    __syncthreads();
    int m = lane & 15, hi = lane >> 4;
    float acc = 0.f;
    int k0 = hi * 72, k1 = hi ? 143 : 72;
    for (int k = k0; k < k1; ++k) acc += (float)sH[wave][m*144 + k] * W2[h*143 + k];
    sPart[wave][hi][m] = acc;
    __syncthreads();
    if (lane < 16) {
      float raw = sPart[wave][0][lane] + sPart[wave][1][lane] + b2[h];
      RAW[(size_t)(e0+lane)*4 + h] = raw;
      atomicMaxF(&M[(size_t)sRow[wave][lane]*4 + h], raw);
    }
    __syncthreads();
  }
}

// ----------------- pass 2: exp + segment sum -------------------------------
__global__ void k_exp(const int* ei, float* RAW, const float* M, float* Z){
  int i = blockIdx.x * blockDim.x + threadIdx.x;
  if (i >= NE*4) return;
  int e = i >> 2, h = i & 3;
  int r = ei[e];
  float ex = __expf(RAW[i] - M[(size_t)r*4 + h]);
  RAW[i] = ex;
  atomicAdd(&Z[(size_t)r*4 + h], ex);
}

// ----------------- pass 3: msg pipeline + normalized scatter ---------------
__global__ void k_edge(const float* s_edge, const float* v_edge, const int* ei,
                       const float* U, const float* RAW, const float* Z,
                       const _Float16* pkEW1, const float* eb1,
                       const _Float16* pkEW2, const float* eb2,
                       const _Float16* pkMW1, const float* mb1,
                       const _Float16* pkMW2, const float* mb2,
                       const _Float16* pkSW1, const float* sb1,
                       const _Float16* pkSW2, const float* sb2pad,
                       float* SAGG, float* VAGG)
{
  __shared__ _Float16 sA[2][16*384];
  __shared__ _Float16 sB[2][16*256];
  __shared__ float    sF[2][16*128];
  __shared__ float    sS[2][16*16];
  __shared__ float    sAttn[2][16], sCoef[2][16];
  __shared__ int      sRow[2][16], sCol[2][16];
  int wave = threadIdx.x >> 5, lane = threadIdx.x & 31;
  int e0 = (blockIdx.x * 2 + wave) * 16;
  if (lane < 16) { sRow[wave][lane] = ei[e0+lane]; sCol[wave][lane] = ei[NE + e0 + lane]; }
  __syncthreads();
  if (lane < 16) {
    int e = e0 + lane, r = sRow[wave][lane];
    float a = 0.f;
    for (int h = 0; h < 4; ++h) a += RAW[(size_t)e*4 + h] * fastrcp(Z[(size_t)r*4 + h]);
    sAttn[wave][lane] = 0.25f * a;
  }
  _Float16* A = sA[wave]; _Float16* B = sB[wave]; float* F = sF[wave];
  // edge-feature staging: [s_edge | v_edge] -> cols 0..142, pad to 160
  for (int idx = lane; idx < 16*160; idx += 32) {
    int m = idx / 160, c = idx % 160;
    float val = 0.f;
    if      (c < 128) val = s_edge[(size_t)(e0+m)*128 + c];
    else if (c < 143) val = v_edge[(size_t)(e0+m)*15 + (c-128)];
    A[m*384 + c] = (_Float16)val;
  }
  __syncthreads();
  wmma_gemm16(A, 384, 5, pkEW1, 16, 0, 16, eb1, true, B, 256, nullptr, 0);   // edge L1
  __syncthreads();
  wmma_gemm16(B, 256, 8, pkEW2, 8, 0, 8, eb2, false, A + 256, 384, nullptr, 0); // u_e -> cols 256..383
  __syncthreads();
  // gather u_i, u_j into cols 0..255
  for (int idx = lane; idx < 16*256; idx += 32) {
    int m = idx >> 8, c = idx & 255;
    float val = (c < 128) ? U[(size_t)sRow[wave][m]*128 + c]
                          : U[(size_t)sCol[wave][m]*128 + (c-128)];
    A[m*384 + c] = (_Float16)val;
  }
  __syncthreads();
  wmma_gemm16(A, 384, 12, pkMW1, 16, 0, 16, mb1, true, B, 256, nullptr, 0);  // msg L1
  __syncthreads();
  wmma_gemm16(B, 256, 8, pkMW2, 8, 0, 8, mb2, false, nullptr, 0, F, 128);    // msg (f32)
  __syncthreads();
  for (int idx = lane; idx < 16*128; idx += 32) {                            // msg -> f16 staging
    int m = idx >> 7, c = idx & 127;
    A[m*384 + c] = (_Float16)F[m*128 + c];
  }
  __syncthreads();
  wmma_gemm16(A, 384, 4, pkSW1, 4, 0, 4, sb1, true, B, 256, nullptr, 0);     // s2v L1 (64)
  __syncthreads();
  wmma_gemm16(B, 256, 2, pkSW2, 1, 0, 1, sb2pad, false, nullptr, 0, sS[wave], 16); // coeff
  __syncthreads();
  if (lane < 16) sCoef[wave][lane] = sS[wave][lane*16];
  __syncthreads();
  for (int idx = lane; idx < 16*128; idx += 32) {
    int m = idx >> 7, c = idx & 127;
    atomicAdd(&SAGG[(size_t)sRow[wave][m]*128 + c], sAttn[wave][m] * F[m*128 + c]);
  }
  for (int idx = lane; idx < 16*15; idx += 32) {
    int m = idx / 15, d = idx % 15;
    atomicAdd(&VAGG[(size_t)sRow[wave][m]*15 + d],
              sAttn[wave][m] * sCoef[wave][m] * v_edge[(size_t)(e0+m)*15 + d]);
  }
}

// ----------------- pass 4: node updates ------------------------------------
__global__ void k_node2(const float* s, const float* v, const float* s_cls,
                        const float* v_cls, const int* batch,
                        const float* SAGG, const float* VAGG,
                        const _Float16* pkDW1, const float* db1,
                        const _Float16* pkDW2, const float* db2,
                        const float* vdW1, const float* vdb1,
                        const float* vdW2, const float* vdb2,
                        const _Float16* pkGW1, const float* gb1,
                        const _Float16* pkGW2, const float* gb2pad,
                        float* X, float* SUMX, float* CNT,
                        float* SES, float* VNS, float* VES, float* out_v)
{
  __shared__ _Float16 sA[2][16*384];
  __shared__ _Float16 sB[2][16*128];
  __shared__ float    sF[2][16*128];
  __shared__ float    sS[2][16*16];
  __shared__ float    sVd[2][16][15];
  __shared__ float    sNrm[2][16][3];
  __shared__ float    sGate[2][16];
  __shared__ int      sG[2][16];
  int wave = threadIdx.x >> 5, lane = threadIdx.x & 31;
  int n0 = (blockIdx.x * 2 + wave) * 16;
  if (lane < 16) sG[wave][lane] = batch[n0 + lane];
  __syncthreads();
  _Float16* A = sA[wave]; _Float16* B = sB[wave]; float* F = sF[wave];
  for (int idx = lane; idx < 16*384; idx += 32) {
    int m = idx / 384, c = idx % 384; int node = n0 + m; int g = sG[wave][m];
    float val;
    if      (c < 128) val = s[(size_t)node*128 + c];
    else if (c < 256) val = SAGG[(size_t)node*128 + (c-128)];
    else              val = s_cls[(size_t)g*128 + (c-256)];
    A[m*384 + c] = (_Float16)val;
  }
  __syncthreads();
  wmma_gemm16(A, 384, 12, pkDW1, 8, 0, 8, db1, true, B, 128, nullptr, 0);
  __syncthreads();
  wmma_gemm16(B, 128, 4, pkDW2, 8, 0, 8, db2, false, nullptr, 0, F, 128);
  __syncthreads();
  for (int idx = lane; idx < 16*128; idx += 32) {
    int m = idx >> 7, c = idx & 127; int node = n0 + m; int g = sG[wave][m];
    float xv = s[(size_t)node*128 + c] + F[m*128 + c];
    X[(size_t)node*128 + c] = xv;
    atomicAdd(&SUMX[(size_t)g*128 + c], xv);
    atomicAdd(&SES[(size_t)g*128 + c], SAGG[(size_t)node*128 + c]);
  }
  if (lane < 16) atomicAdd(&CNT[sG[wave][lane]], 1.f);
  // v_delta (45->15->15), norms and cosine: one lane per node
  if (lane < 16) {
    int node = n0 + lane; int g = sG[wave][lane];
    float in[45];
    for (int k = 0; k < 15; ++k) {
      in[k]      = v[(size_t)node*15 + k];
      in[15 + k] = VAGG[(size_t)node*15 + k];
      in[30 + k] = v_cls[(size_t)g*15 + k];
    }
    float hdn[15];
    for (int o = 0; o < 15; ++o) {
      float a = vdb1[o];
      for (int k = 0; k < 45; ++k) a += in[k] * vdW1[k*15 + o];
      hdn[o] = siluf(a);
    }
    float vn2 = 0.f, dn2 = 0.f, dot = 0.f;
    for (int o = 0; o < 15; ++o) {
      float a = vdb2[o];
      for (int k = 0; k < 15; ++k) a += hdn[k] * vdW2[k*15 + o];
      sVd[wave][lane][o] = a;
      dn2 += a * a; dot += in[o] * a; vn2 += in[o] * in[o];
    }
    float vn = sqrtf(vn2), dn = sqrtf(dn2);
    sNrm[wave][lane][0] = vn; sNrm[wave][lane][1] = dn;
    sNrm[wave][lane][2] = dot / (vn * dn + 1e-6f);
  }
  __syncthreads();
  // gate staging: [v_norm, d_norm, cos, s] -> 131, pad 160
  for (int idx = lane; idx < 16*160; idx += 32) {
    int m = idx / 160, c = idx % 160; int node = n0 + m;
    float val = 0.f;
    if      (c < 3)   val = sNrm[wave][m][c];
    else if (c < 131) val = s[(size_t)node*128 + (c-3)];
    A[m*384 + c] = (_Float16)val;
  }
  __syncthreads();
  wmma_gemm16(A, 384, 5, pkGW1, 8, 0, 8, gb1, true, B, 128, nullptr, 0);
  __syncthreads();
  wmma_gemm16(B, 128, 4, pkGW2, 1, 0, 1, gb2pad, false, nullptr, 0, sS[wave], 16);
  __syncthreads();
  if (lane < 16) sGate[wave][lane] = sigmf(sS[wave][lane*16]);
  __syncthreads();
  for (int idx = lane; idx < 16*15; idx += 32) {
    int m = idx / 15, d = idx % 15; int node = n0 + m; int g = sG[wave][m];
    float vo = v[(size_t)node*15 + d] + sVd[wave][m][d] * sGate[wave][m];
    out_v[(size_t)node*15 + d] = vo;
    atomicAdd(&VNS[(size_t)g*15 + d], vo);
    atomicAdd(&VES[(size_t)g*15 + d], VAGG[(size_t)node*15 + d]);
  }
}

// ----------------- GraphNorm passes ----------------------------------------
__global__ void k_gmean(float* SUMX, const float* CNT){
  int i = blockIdx.x * blockDim.x + threadIdx.x;
  if (i >= NB*128) return;
  SUMX[i] /= fmaxf(CNT[i >> 7], 1.f);
}

__global__ void k_var1(const int* batch, float* X, const float* MEAN,
                       const float* gms, float* VARS){
  int i = blockIdx.x * blockDim.x + threadIdx.x;
  if (i >= NN*128) return;
  int n = i >> 7, c = i & 127;
  int g = batch[n];
  float xc = X[i] - gms[c] * MEAN[(size_t)g*128 + c];
  X[i] = xc;
  atomicAdd(&VARS[(size_t)g*128 + c], xc * xc);
}

__global__ void k_var2(float* VARS, const float* CNT){
  int i = blockIdx.x * blockDim.x + threadIdx.x;
  if (i >= NB*128) return;
  VARS[i] = rsqrtf(VARS[i] / fmaxf(CNT[i >> 7], 1.f) + 1e-5f);
}

__global__ void k_sout(const int* batch, const float* X, const float* INV,
                       const float* gnw, const float* gnb,
                       float* out_s, float* SNS){
  int i = blockIdx.x * blockDim.x + threadIdx.x;
  if (i >= NN*128) return;
  int n = i >> 7, c = i & 127;
  int g = batch[n];
  float so = gnw[c] * X[i] * INV[(size_t)g*128 + c] + gnb[c];
  out_s[i] = so;
  atomicAdd(&SNS[(size_t)g*128 + c], so);
}

// ----------------- pass 7: CLS updates -------------------------------------
__global__ void k_cls(const float* s_cls, const float* v_cls,
                      const float* SNS, const float* SES,
                      const float* VNS, const float* VES, const float* CNT,
                      const _Float16* pkW1, const float* b1,
                      const _Float16* pkW2, const float* b2,
                      const float* vcW1, const float* vcb1,
                      const float* vcW2, const float* vcb2,
                      const float* lnw, const float* lnb,
                      float* out_sc, float* out_vc)
{
  __shared__ _Float16 sA[2][16*384];
  __shared__ _Float16 sB[2][16*128];
  __shared__ float    sF[2][16*128];
  __shared__ float    sMu[2][16], sIv[2][16];
  int wave = threadIdx.x >> 5, lane = threadIdx.x & 31;
  int g0 = (blockIdx.x * 2 + wave) * 16;
  _Float16* A = sA[wave]; float* F = sF[wave];
  for (int idx = lane; idx < 16*384; idx += 32) {
    int m = idx / 384, c = idx % 384; int g = g0 + m;
    float cnt = fmaxf(CNT[g], 1.f);
    float val;
    if      (c < 128) val = s_cls[(size_t)g*128 + c];
    else if (c < 256) val = SNS[(size_t)g*128 + (c-128)] / cnt;
    else              val = SES[(size_t)g*128 + (c-256)] / cnt;
    A[m*384 + c] = (_Float16)val;
  }
  __syncthreads();
  wmma_gemm16(A, 384, 12, pkW1, 8, 0, 8, b1, true, sB[wave], 128, nullptr, 0);
  __syncthreads();
  wmma_gemm16(sB[wave], 128, 4, pkW2, 8, 0, 8, b2, false, nullptr, 0, F, 128);
  __syncthreads();
  for (int idx = lane; idx < 16*128; idx += 32) {
    int m = idx >> 7, c = idx & 127;
    F[m*128 + c] += s_cls[(size_t)(g0+m)*128 + c];     // yc
  }
  __syncthreads();
  if (lane < 16) {
    float mu = 0.f;
    for (int c = 0; c < 128; ++c) mu += F[lane*128 + c];
    mu *= (1.f/128.f);
    float vv = 0.f;
    for (int c = 0; c < 128; ++c) { float d = F[lane*128 + c] - mu; vv += d*d; }
    vv *= (1.f/128.f);
    sMu[wave][lane] = mu; sIv[wave][lane] = rsqrtf(vv + 1e-5f);
  }
  __syncthreads();
  for (int idx = lane; idx < 16*128; idx += 32) {
    int m = idx >> 7, c = idx & 127;
    out_sc[(size_t)(g0+m)*128 + c] =
        lnw[c] * (F[m*128 + c] - sMu[wave][m]) * sIv[wave][m] + lnb[c];
  }
  // v_cls update (45->15->15)
  if (lane < 16) {
    int g = g0 + lane;
    float cnt = fmaxf(CNT[g], 1.f);
    float in[45];
    for (int k = 0; k < 15; ++k) {
      in[k]      = v_cls[(size_t)g*15 + k];
      in[15 + k] = VNS[(size_t)g*15 + k] / cnt;
      in[30 + k] = VES[(size_t)g*15 + k] / cnt;
    }
    float hdn[15];
    for (int o = 0; o < 15; ++o) {
      float a = vcb1[o];
      for (int k = 0; k < 45; ++k) a += in[k] * vcW1[k*15 + o];
      hdn[o] = siluf(a);
    }
    for (int o = 0; o < 15; ++o) {
      float a = vcb2[o];
      for (int k = 0; k < 15; ++k) a += hdn[k] * vcW2[k*15 + o];
      out_vc[(size_t)g*15 + o] = in[o] + a;
    }
  }
}

// ---------------------------------------------------------------------------
extern "C" void kernel_launch(void* const* d_in, const int* in_sizes, int n_in,
                              void* d_out, int out_size, void* d_ws, size_t ws_size,
                              hipStream_t stream)
{
  (void)in_sizes; (void)n_in; (void)out_size; (void)ws_size;
  const float* s      = (const float*)d_in[0];
  const float* v      = (const float*)d_in[1];
  const float* s_edge = (const float*)d_in[2];
  const float* v_edge = (const float*)d_in[3];
  const int*   ei     = (const int*)d_in[4];
  const float* s_cls  = (const float*)d_in[5];
  const float* v_cls  = (const float*)d_in[6];
  const int*   batch  = (const int*)d_in[7];
  // params flattened in dict/list order
  const float* nodeW1 = (const float*)d_in[8];  const float* nodeB1 = (const float*)d_in[9];
  const float* nodeW2 = (const float*)d_in[10]; const float* nodeB2 = (const float*)d_in[11];
  const float* edgeW1 = (const float*)d_in[12]; const float* edgeB1 = (const float*)d_in[13];
  const float* edgeW2 = (const float*)d_in[14]; const float* edgeB2 = (const float*)d_in[15];
  const float* msgW1  = (const float*)d_in[16]; const float* msgB1  = (const float*)d_in[17];
  const float* msgW2  = (const float*)d_in[18]; const float* msgB2  = (const float*)d_in[19];
  const float* s2vW1  = (const float*)d_in[20]; const float* s2vB1  = (const float*)d_in[21];
  const float* s2vW2  = (const float*)d_in[22]; const float* s2vB2  = (const float*)d_in[23];
  const float* attnW1 = (const float*)d_in[24]; const float* attnB1 = (const float*)d_in[25];
  const float* attnW2 = (const float*)d_in[26]; const float* attnB2 = (const float*)d_in[27];
  const float* sdW1   = (const float*)d_in[28]; const float* sdB1   = (const float*)d_in[29];
  const float* sdW2   = (const float*)d_in[30]; const float* sdB2   = (const float*)d_in[31];
  const float* vdW1   = (const float*)d_in[32]; const float* vdB1   = (const float*)d_in[33];
  const float* vdW2   = (const float*)d_in[34]; const float* vdB2   = (const float*)d_in[35];
  const float* gnw    = (const float*)d_in[36];
  const float* gnb    = (const float*)d_in[37];
  const float* gms    = (const float*)d_in[38];
  const float* gtW1   = (const float*)d_in[39]; const float* gtB1   = (const float*)d_in[40];
  const float* gtW2   = (const float*)d_in[41]; const float* gtB2   = (const float*)d_in[42];
  const float* scW1   = (const float*)d_in[43]; const float* scB1   = (const float*)d_in[44];
  const float* scW2   = (const float*)d_in[45]; const float* scB2   = (const float*)d_in[46];
  const float* vcW1   = (const float*)d_in[47]; const float* vcB1   = (const float*)d_in[48];
  const float* vcW2   = (const float*)d_in[49]; const float* vcB2   = (const float*)d_in[50];
  const float* lnw    = (const float*)d_in[51];
  const float* lnb    = (const float*)d_in[52];

  float* out_s  = (float*)d_out;
  float* out_v  = out_s  + (size_t)NN * 128;
  float* out_sc = out_v  + (size_t)NN * 15;
  float* out_vc = out_sc + (size_t)NB * 128;

  // workspace carve-out (float units, 256B-aligned slabs)
  float* w = (float*)d_ws;
  size_t off = 0;
  auto take = [&](size_t n){ float* p = w + off; off += (n + 63) & ~(size_t)63; return p; };
  float* M    = take((size_t)NN * 4);
  float* zb0  = w + off;                // --- zero-initialized block begins
  float* Z    = take((size_t)NN * 4);
  float* SAGG = take((size_t)NN * 128);
  float* VAGG = take((size_t)NN * 15);
  float* SUMX = take((size_t)NB * 128);
  float* VARS = take((size_t)NB * 128);
  float* CNT  = take((size_t)NB);
  float* SNS  = take((size_t)NB * 128);
  float* SES  = take((size_t)NB * 128);
  float* VNS  = take((size_t)NB * 15);
  float* VES  = take((size_t)NB * 15);
  size_t zeroN = (size_t)((w + off) - zb0);  // --- zero block ends
  float* U    = take((size_t)NN * 128);
  float* RAW  = take((size_t)NE * 4);
  float* X    = take((size_t)NN * 128);
  float* b_s2v2  = take(16);
  float* b_gate2 = take(16);
  float* b_attn1 = take(576);
  _Float16* hp = (_Float16*)(w + off);
  size_t hoff = 0;
  auto takeh = [&](size_t n){ _Float16* p = hp + hoff; hoff += (n + 63) & ~(size_t)63; return p; };
  _Float16* pkNode1 = takeh(5  * 16 * 512);
  _Float16* pkNode2 = takeh(8  * 8  * 512);
  _Float16* pkEdge1 = takeh(5  * 16 * 512);
  _Float16* pkEdge2 = takeh(8  * 8  * 512);
  _Float16* pkMsg1  = takeh(12 * 16 * 512);
  _Float16* pkMsg2  = takeh(8  * 8  * 512);
  _Float16* pkS2v1  = takeh(4  * 4  * 512);
  _Float16* pkS2v2  = takeh(2  * 1  * 512);
  _Float16* pkAttn1 = takeh(14 * 36 * 512);
  _Float16* pkSd1   = takeh(12 * 8  * 512);
  _Float16* pkSd2   = takeh(4  * 8  * 512);
  _Float16* pkGt1   = takeh(5  * 8  * 512);
  _Float16* pkGt2   = takeh(4  * 1  * 512);
  _Float16* pkSc1   = takeh(12 * 8  * 512);
  _Float16* pkSc2   = takeh(4  * 8  * 512);

  auto FILL = [&](float* p, float val, size_t n){
    k_fill<<<dim3((unsigned)((n + 255) / 256)), dim3(256), 0, stream>>>(p, val, (int)n);
  };
  auto PACK = [&](const float* W, _Float16* dst, int K, int N, int ldw,
                  int KT, int NTtot, int nt0, int NTcnt){
    int total = KT * NTcnt * 512;
    k_pack<<<dim3((total + 255) / 256), dim3(256), 0, stream>>>(W, dst, K, N, ldw, KT, NTtot, nt0, NTcnt);
  };

  // init accumulators
  FILL(M, -3.4e38f, (size_t)NN * 4);
  FILL(zb0, 0.f, zeroN);
  // padded biases
  k_padbias<<<1, 32, 0, stream>>>(s2vB2, b_s2v2, 1, 16);
  k_padbias<<<1, 32, 0, stream>>>(gtB2, b_gate2, 1, 16);
  for (int h = 0; h < 4; ++h)
    k_padbias<<<1, 160, 0, stream>>>(attnB1 + h*143, b_attn1 + h*144, 143, 144);
  // pack weights into WMMA B-fragment layout
  PACK(nodeW1, pkNode1, 143, 256, 256, 5, 16, 0, 16);
  PACK(nodeW2, pkNode2, 256, 128, 128, 8,  8, 0,  8);
  PACK(edgeW1, pkEdge1, 143, 256, 256, 5, 16, 0, 16);
  PACK(edgeW2, pkEdge2, 256, 128, 128, 8,  8, 0,  8);
  PACK(msgW1,  pkMsg1,  384, 256, 256, 12,16, 0, 16);
  PACK(msgW2,  pkMsg2,  256, 128, 128, 8,  8, 0,  8);
  PACK(s2vW1,  pkS2v1,  128,  64,  64, 4,  4, 0,  4);
  PACK(s2vW2,  pkS2v2,   64,   1,   1, 2,  1, 0,  1);
  for (int h = 0; h < 4; ++h)
    PACK(attnW1 + (size_t)h*429*143, pkAttn1, 429, 143, 143, 14, 36, 9*h, 9);
  PACK(sdW1,   pkSd1,   384, 128, 128, 12, 8, 0,  8);
  PACK(sdW2,   pkSd2,   128, 128, 128, 4,  8, 0,  8);
  PACK(gtW1,   pkGt1,   131, 128, 128, 5,  8, 0,  8);
  PACK(gtW2,   pkGt2,   128,   1,   1, 4,  1, 0,  1);
  PACK(scW1,   pkSc1,   384, 128, 128, 12, 8, 0,  8);
  PACK(scW2,   pkSc2,   128, 128, 128, 4,  8, 0,  8);

  // pass 0: per-node MLP (computed once per node instead of twice per edge)
  k_node_u<<<dim3(NN/32), dim3(64), 0, stream>>>(s, v, pkNode1, nodeB1, pkNode2, nodeB2, U);
  // pass 1: attention logits + segment max
  k_attn<<<dim3(NE/32), dim3(64), 0, stream>>>(s, v, s_edge, v_edge, ei,
                                               pkAttn1, b_attn1, attnW2, attnB2, RAW, M);
  // pass 2: exp + segment sum
  k_exp<<<dim3((NE*4 + 255)/256), dim3(256), 0, stream>>>(ei, RAW, M, Z);
  // pass 3: msg pipeline + normalized scatter-aggregation
  k_edge<<<dim3(NE/32), dim3(64), 0, stream>>>(s_edge, v_edge, ei, U, RAW, Z,
                                               pkEdge1, edgeB1, pkEdge2, edgeB2,
                                               pkMsg1, msgB1, pkMsg2, msgB2,
                                               pkS2v1, s2vB1, pkS2v2, b_s2v2,
                                               SAGG, VAGG);
  // pass 4: node updates (s_delta, v_delta, gate, v_out, per-graph sums)
  k_node2<<<dim3(NN/32), dim3(64), 0, stream>>>(s, v, s_cls, v_cls, batch, SAGG, VAGG,
                                                pkSd1, sdB1, pkSd2, sdB2,
                                                vdW1, vdB1, vdW2, vdB2,
                                                pkGt1, gtB1, pkGt2, b_gate2,
                                                X, SUMX, CNT, SES, VNS, VES, out_v);
  // GraphNorm
  k_gmean<<<dim3((NB*128 + 255)/256), dim3(256), 0, stream>>>(SUMX, CNT);
  k_var1<<<dim3((NN*128 + 255)/256), dim3(256), 0, stream>>>(batch, X, SUMX, gms, VARS);
  k_var2<<<dim3((NB*128 + 255)/256), dim3(256), 0, stream>>>(VARS, CNT);
  k_sout<<<dim3((NN*128 + 255)/256), dim3(256), 0, stream>>>(batch, X, VARS, gnw, gnb, out_s, SNS);
  // CLS updates
  k_cls<<<dim3(NB/32), dim3(64), 0, stream>>>(s_cls, v_cls, SNS, SES, VNS, VES, CNT,
                                              pkSc1, scB1, pkSc2, scB2,
                                              vcW1, vcB1, vcW2, vcB2,
                                              lnw, lnb, out_sc, out_vc);
}
